// AttentionLayer_51994874085709
// MI455X (gfx1250) — matrix operand order repaired and checked by reading, not compiled, not run on hardware
//
#include <hip/hip_runtime.h>
#include <cstdint>
#include <cstddef>

// ---------------------------------------------------------------------------
// GAT layer for MI455X (gfx1250, wave32, WMMA + async-to-LDS).
//   h = x @ W^T                      (WMMA bf16, f32 accum)
//   e = leakyrelu(f1_i + f2_j) * adj ; e==0 -> -1e9 ; softmax rows ; elu(alpha@h)
// Single-pass fused softmax-matmul using analytic per-row max bound
//   M_i = leakyrelu(f1_i + max_j f2_j)   (exact: softmax is shift-invariant)
// HBM roofline: adj = 256 MB read once ~= 11 us @ 23.3 TB/s; h (4 MB bf16)
// lives in L2 and is streamed to LDS via GLOBAL_LOAD_ASYNC_TO_LDS_B128 with
// double buffering so the fill overlaps the WMMA stream.
// ---------------------------------------------------------------------------

typedef __attribute__((ext_vector_type(16))) __bf16 v16bf;
typedef __attribute__((ext_vector_type(8)))  __bf16 v8bf;
typedef __attribute__((ext_vector_type(8)))  float  v8f;

#define GAT_N     8192
#define GAT_DIN   512
#define GAT_DOUT  256
#define GAT_NEGINF (-1e9f)

// -------------------------------- fp32 -> bf16 convert ----------------------
__global__ void gat_cvt_bf16(const float* __restrict__ s, __bf16* __restrict__ d, int n) {
    int i = blockIdx.x * blockDim.x + threadIdx.x;
    if (i < n) d[i] = (__bf16)s[i];
}

// -------------------------------- h = x @ W^T -------------------------------
// One 16x16 output tile per wave; K = 512 in chunks of 32.
// A (16x32 bf16) lane layout: lane L holds row M=L%16; element e -> K = k0 + (L/16)*16 + e
// B (32x16 bf16) lane layout: lane L holds col N=L%16; element e -> K = k0 + (L/16)*16 + e
// D (16x16 f32):  vgpr r, lane L -> row = r + (L/16)*8, col = L%16
__global__ void gat_h_gemm(const __bf16* __restrict__ Xb, const __bf16* __restrict__ Wb,
                           float* __restrict__ h, __bf16* __restrict__ hTt) {
    const int tid  = threadIdx.x;
    const int lane = tid & 31;
    const int wv   = tid >> 5;
    const int gw   = blockIdx.x * 8 + wv;        // 0 .. 8191  (512 mtiles x 16 ntiles)
    const int mb   = (gw >> 4) * 16;
    const int nb   = (gw & 15) * 16;
    const int l16  = lane & 15;
    const int half = lane >> 4;

    const __bf16* arow = Xb + (size_t)(mb + l16) * GAT_DIN;
    const __bf16* brow = Wb + (size_t)(nb + l16) * GAT_DIN + half * 16;

    v8f acc = {0.f, 0.f, 0.f, 0.f, 0.f, 0.f, 0.f, 0.f};

    #pragma unroll
    for (int k0 = 0; k0 < GAT_DIN; k0 += 32) {
        v8bf lo = *(const v8bf*)(arow + k0 + half * 16);
        v8bf hi = *(const v8bf*)(arow + k0 + half * 16 + 8);
        v16bf A;
        #pragma unroll
        for (int e = 0; e < 8; ++e) { A[e] = lo[e]; A[e + 8] = hi[e]; }
        v16bf B = *(const v16bf*)(brow + k0);
        acc = __builtin_amdgcn_wmma_f32_16x16x32_bf16(false, A, false, B,
                                                      (short)0, acc, false, false);
    }

    #pragma unroll
    for (int r = 0; r < 8; ++r) {
        const int row = mb + r + half * 8;
        const int col = nb + l16;
        const float v = acc[r];
        h[(size_t)row * GAT_DOUT + col] = v;
        // chunk-tiled transposed bf16: [chunk=row/32][col][row%32] -> contiguous 16KB per chunk
        hTt[(size_t)(row >> 5) * (GAT_DOUT * 32) + col * 32 + (row & 31)] = (__bf16)v;
    }
}

// ------------------------ f1 = h@a1, f2 = h@a2, blockmax(f2) ----------------
__global__ void gat_f(const float* __restrict__ h, const float* __restrict__ a1,
                      const float* __restrict__ a2, float* __restrict__ f1,
                      float* __restrict__ f2, float* __restrict__ blockmax) {
    __shared__ float smax[8];
    const int tid  = threadIdx.x;
    const int lane = tid & 31;
    const int wv   = tid >> 5;
    const int row  = blockIdx.x * 8 + wv;

    float s1 = 0.f, s2 = 0.f;
    #pragma unroll
    for (int t = 0; t < 8; ++t) {
        const int c = t * 32 + lane;
        const float hv = h[(size_t)row * GAT_DOUT + c];
        s1 = fmaf(hv, a1[c], s1);
        s2 = fmaf(hv, a2[c], s2);
    }
    #pragma unroll
    for (int off = 16; off > 0; off >>= 1) {
        s1 += __shfl_xor(s1, off, 32);
        s2 += __shfl_xor(s2, off, 32);
    }
    if (lane == 0) {
        f1[row] = s1;
        f2[row] = s2;
        smax[wv] = s2;
    }
    __syncthreads();
    if (tid == 0) {
        float m = smax[0];
        #pragma unroll
        for (int i = 1; i < 8; ++i) m = fmaxf(m, smax[i]);
        blockmax[blockIdx.x] = m;
    }
}

__global__ void gat_f2max(const float* __restrict__ blockmax, float* __restrict__ f2max) {
    __shared__ float sm[256];
    const int tid = threadIdx.x;
    float m = -3.4e38f;
    for (int i = tid; i < 1024; i += 256) m = fmaxf(m, blockmax[i]);
    sm[tid] = m;
    __syncthreads();
    for (int off = 128; off > 0; off >>= 1) {
        if (tid < off) sm[tid] = fmaxf(sm[tid], sm[tid + off]);
        __syncthreads();
    }
    if (tid == 0) f2max[0] = sm[0];
}

// --------------- fused masked-softmax @ h (flash-style, single pass) --------
// 4 waves/block, 16 rows/wave (64 rows/block), 128 blocks.
// Loop over 256 chunks of 32 columns; h tile (32x256 bf16 = 16KB) double-
// buffered in LDS via GLOBAL_LOAD_ASYNC_TO_LDS_B128 (ASYNCcnt-tracked) so the
// fetch of chunk jc+1 overlaps probability math + 16 WMMAs on chunk jc.
__global__ void __launch_bounds__(128)
gat_attn(const float* __restrict__ adj, const __bf16* __restrict__ hTt,
         const float* __restrict__ f1, const float* __restrict__ f2,
         const float* __restrict__ f2maxp, float* __restrict__ out) {
    __shared__ __align__(16) __bf16 sB[2][GAT_DOUT * 32];   // 2 x 16 KB

    const int tid  = threadIdx.x;
    const int lane = tid & 31;
    const int wv   = tid >> 5;
    const int l16  = lane & 15;
    const int half = lane >> 4;
    const int rb   = blockIdx.x * 64 + wv * 16;
    const int row  = rb + l16;

    const float f1v = f1[row];
    const float f2m = f2maxp[0];
    const float tb  = f1v + f2m;
    const float Mv  = tb > 0.f ? tb : 0.01f * tb;        // leakyrelu bound on row max

    const float* adjrow = adj + (size_t)row * GAT_N + half * 16;
    const float* f2p    = f2 + half * 16;

    // LDS byte offsets of the two buffers (generic ptr low 32 bits = DS offset)
    const uint32_t lds0 = (uint32_t)(uintptr_t)&sB[0][0];
    const uint32_t lds1 = (uint32_t)(uintptr_t)&sB[1][0];

    // issue one 16KB chunk as 8 per-wave async b128 loads (this thread moves
    // 16B per load; 128 threads x 8 x 16B = 16KB), no VGPR round-trip.
    auto issue_chunk = [&](int jc, uint32_t ldsbase) {
        const uint64_t gbase = (uint64_t)(uintptr_t)(hTt + (size_t)jc * (GAT_DOUT * 32));
        #pragma unroll
        for (int i = 0; i < 8; ++i) {
            const uint32_t boff = (uint32_t)(tid + i * 128) * 16u;
            const uint64_t ga = gbase + boff;
            const uint32_t la = ldsbase + boff;
            asm volatile("global_load_async_to_lds_b128 %0, %1, off"
                         :: "v"(la), "v"(ga) : "memory");
        }
    };

    v8f acc[16];
    #pragma unroll
    for (int nt = 0; nt < 16; ++nt)
        acc[nt] = (v8f){0.f, 0.f, 0.f, 0.f, 0.f, 0.f, 0.f, 0.f};
    float rowsum = 0.f;

    issue_chunk(0, lds0);                                // prologue: chunk 0 -> buf0

    for (int jc = 0; jc < GAT_N / 32; ++jc) {
        const int cur = jc & 1;
        const int j0  = jc * 32;

        if (jc + 1 < GAT_N / 32) {
            issue_chunk(jc + 1, cur ? lds0 : lds1);      // prefetch next chunk
            asm volatile("s_wait_asynccnt 0x8" ::: "memory");  // batch jc landed
        } else {
            asm volatile("s_wait_asynccnt 0x0" ::: "memory");
        }
        __syncthreads();                                 // all waves' fills visible

        __builtin_prefetch(adjrow + j0 + 32, 0, 1);      // next adj chunk -> global_prefetch_b8

        // ---- per-lane logits -> probabilities, directly in A-matrix layout ----
        float __align__(16) ax[16];
        float __align__(16) fx[16];
        *(float4*)&ax[0]  = *(const float4*)(adjrow + j0);
        *(float4*)&ax[4]  = *(const float4*)(adjrow + j0 + 4);
        *(float4*)&ax[8]  = *(const float4*)(adjrow + j0 + 8);
        *(float4*)&ax[12] = *(const float4*)(adjrow + j0 + 12);
        *(float4*)&fx[0]  = *(const float4*)(f2p + j0);
        *(float4*)&fx[4]  = *(const float4*)(f2p + j0 + 4);
        *(float4*)&fx[8]  = *(const float4*)(f2p + j0 + 8);
        *(float4*)&fx[12] = *(const float4*)(f2p + j0 + 12);

        v16bf P;
        #pragma unroll
        for (int e = 0; e < 16; ++e) {
            const float t  = f1v + fx[e];
            const float lk = t > 0.f ? t : 0.01f * t;
            float m = lk * ax[e];
            m = (m == 0.f) ? GAT_NEGINF : m;
            const float p = __expf(m - Mv);              // p <= 1, masked -> 0
            rowsum += p;
            P[e] = (__bf16)p;
        }

        // ---- 16 WMMAs: acc[nt] += P (16x32) @ h_tile (32x16) ----
        const __bf16* sbuf = &sB[cur][0];
        #pragma unroll
        for (int nt = 0; nt < 16; ++nt) {
            const v16bf B = *(const v16bf*)(sbuf + nt * 512 + l16 * 32 + half * 16);
            acc[nt] = __builtin_amdgcn_wmma_f32_16x16x32_bf16(false, P, false, B,
                                                              (short)0, acc[nt], false, false);
        }

        __syncthreads();   // all reads of buf[cur] done before next iter refills it
    }

    // ---- finalize: out = elu(acc / rowsum) ----
    rowsum += __shfl_xor(rowsum, 16, 32);                // lanes l and l+16 cover same row
    const float inv = 1.0f / rowsum;                     // inv for row rb + l16

    #pragma unroll
    for (int nt = 0; nt < 16; ++nt) {
        #pragma unroll
        for (int r = 0; r < 8; ++r) {
            const float iv = __shfl(inv, r + half * 8, 32);
            const float v  = acc[nt][r] * iv;
            const float o  = v > 0.f ? v : (__expf(v) - 1.0f);
            out[(size_t)(rb + r + half * 8) * GAT_DOUT + nt * 16 + l16] = o;
        }
    }
}

// ---------------------------------------------------------------------------
extern "C" void kernel_launch(void* const* d_in, const int* in_sizes, int n_in,
                              void* d_out, int out_size, void* d_ws, size_t ws_size,
                              hipStream_t stream) {
    const float* x   = (const float*)d_in[0];   // [8192, 512]
    const float* adj = (const float*)d_in[1];   // [8192, 8192]
    const float* W   = (const float*)d_in[2];   // [256, 512]
    const float* a1  = (const float*)d_in[3];   // [256]
    const float* a2  = (const float*)d_in[4];   // [256]
    float* out = (float*)d_out;                 // [8192, 256] fp32

    (void)in_sizes; (void)n_in; (void)out_size; (void)ws_size;

    // workspace carve-up (256B aligned)
    char* ws = (char*)d_ws;
    size_t off = 0;
    auto alloc = [&](size_t bytes) -> char* {
        char* p = ws + off;
        off += (bytes + 255) & ~(size_t)255;
        return p;
    };
    __bf16* xb   = (__bf16*)alloc((size_t)GAT_N * GAT_DIN * 2);      // 8 MB
    __bf16* wb   = (__bf16*)alloc((size_t)GAT_DOUT * GAT_DIN * 2);   // 256 KB
    __bf16* hTt  = (__bf16*)alloc((size_t)GAT_N * GAT_DOUT * 2);     // 4 MB (chunk-tiled h^T)
    float*  h    = (float*) alloc((size_t)GAT_N * GAT_DOUT * 4);     // 8 MB
    float*  f1   = (float*) alloc((size_t)GAT_N * 4);
    float*  f2   = (float*) alloc((size_t)GAT_N * 4);
    float*  bmax = (float*) alloc(1024 * 4);
    float*  f2mx = (float*) alloc(256);

    const int nx = GAT_N * GAT_DIN;        // 4,194,304
    const int nw = GAT_DOUT * GAT_DIN;     // 131,072
    gat_cvt_bf16<<<(nx + 255) / 256, 256, 0, stream>>>(x, xb, nx);
    gat_cvt_bf16<<<(nw + 255) / 256, 256, 0, stream>>>(W, wb, nw);

    gat_h_gemm<<<1024, 256, 0, stream>>>(xb, wb, h, hTt);            // 8192 tiles, 8 waves/block
    gat_f<<<1024, 256, 0, stream>>>(h, a1, a2, f1, f2, bmax);
    gat_f2max<<<1, 256, 0, stream>>>(bmax, f2mx);
    gat_attn<<<128, 128, 0, stream>>>(adj, hTt, f1, f2, f2mx, out);
}
